// QuantDeepseekMLP_4200478015711
// MI455X (gfx1250) — compile-verified, matrix operand order checked by reading, not run on hardware
//
#include <hip/hip_runtime.h>
#include <stdint.h>

// ---------------------------------------------------------------- types
typedef __attribute__((ext_vector_type(16))) __bf16 v16bf;
typedef __attribute__((ext_vector_type(8)))  __bf16 v8bf;
typedef __attribute__((ext_vector_type(8)))  float  v8f;
typedef __attribute__((ext_vector_type(4)))  float  v4f;
typedef __attribute__((ext_vector_type(4)))  int    v4i;

// ---------------------------------------------------------------- problem dims
#define DIM_H 4096
#define DIM_I 11008
#define DIM_BS 8192          // 4 * 2048
#define NF_BLOCK 64

// ---------------------------------------------------------------- tiling
#define BM 128               // workgroup rows
#define BN 64                // workgroup cols
#define BK 32                // k-step (WMMA bf16 depth)
#define LDSTRIDE 40          // bf16 elems per LDS row (32 + 8 pad; keeps 16B align)

// ---------------------------------------------------------------- CDNA5 async path (guarded)
#if defined(__gfx1250__) && __has_builtin(__builtin_amdgcn_global_load_async_to_lds_b128) && __has_builtin(__builtin_amdgcn_s_wait_asynccnt)
#define USE_ASYNC 1
#else
#define USE_ASYNC 0
#endif

// ---------------------------------------------------------------- NF4 codebook
__device__ __constant__ float NF4_TAB[16] = {
    -1.0f, -0.6961928009986877f, -0.5250730514526367f, -0.39491748809814453f,
    -0.28444138169288635f, -0.18477343022823334f, -0.09105003625154495f, 0.0f,
    0.07958029955625534f, 0.16093020141124725f, 0.24611230194568634f, 0.33791524171829224f,
    0.44070982933044434f, 0.5626170039176941f, 0.7229568362236023f, 1.0f };

// native bf16 convert (RNE fptrunc -> hardware cvt on gfx1250)
__device__ static inline __bf16 f2bf(float f) { return (__bf16)f; }

// Build a 16x32 bf16 A/B fragment from an LDS row-major [row][k] tile.
// ISA 7.12.2: both lane halves carry row = lane&15; vector elems 0..7 =
// K (lane>>4)*8..+7, elems 8..15 = K 16+(lane>>4)*8..+7.
__device__ static inline v16bf load_frag(const unsigned short* base, int row, int lane_hi) {
    const unsigned short* p = base + row * LDSTRIDE + lane_hi * 8;
    v8bf lo = *(const v8bf*)(p);
    v8bf hi = *(const v8bf*)(p + 16);
    return __builtin_shufflevector(lo, hi, 0,1,2,3,4,5,6,7,8,9,10,11,12,13,14,15);
}

// =================================================================
// Kernel 1: inter[bs,i] = silu(x @ Wg^T) * (x @ Wu^T)   (bf16 out)
// software-pipelined: loads(k+1) issued under WMMA(k), double-buffered LDS
// =================================================================
struct Raw1 { v4f x0, x1, x2, x3; v4i g0, g1, u0, u1; float gs, us; };

__device__ static inline Raw1 fetch1(const float* xrow, const int* grow, const int* urow,
                                     const float* gsrow, const float* usrow,
                                     int k0, int a_k, int b_k) {
    Raw1 r;
    const float* s = xrow + k0 + a_k;
    r.x0 = *(const v4f*)(s);
    r.x1 = *(const v4f*)(s + 4);
    r.x2 = *(const v4f*)(s + 8);
    r.x3 = *(const v4f*)(s + 12);
    const int kk = k0 + b_k;
    r.g0 = *(const v4i*)(grow + kk);
    r.g1 = *(const v4i*)(grow + kk + 4);
    r.u0 = *(const v4i*)(urow + kk);
    r.u1 = *(const v4i*)(urow + kk + 4);
    r.gs = gsrow[k0 >> 6];
    r.us = usrow[k0 >> 6];
    return r;
}

__global__ __launch_bounds__(256) void k_gateup(
    const float* __restrict__ x,
    const int*   __restrict__ gcodes, const float* __restrict__ gscales,
    const int*   __restrict__ ucodes, const float* __restrict__ uscales,
    unsigned short* __restrict__ inter)
{
    __shared__ alignas(16) unsigned short sA[2][BM * LDSTRIDE];
    __shared__ alignas(16) unsigned short sG[2][BN * LDSTRIDE];
    __shared__ alignas(16) unsigned short sU[2][BN * LDSTRIDE];
    __shared__ float sNF4[16];

    const int tid = threadIdx.x;
    if (tid < 16) sNF4[tid] = NF4_TAB[tid];

    const int row0 = blockIdx.y * BM;
    const int col0 = blockIdx.x * BN;

    const int lane = tid & 31;
    const int wave = tid >> 5;          // 8 wave32s
    const int mw = wave >> 2;           // 0..1  -> 64-row half
    const int nw = wave & 3;            // 0..3  -> 16-col slice
    const int ln = lane & 15;
    const int lh = lane >> 4;

    // staging assignments
    const int a_m = tid >> 1;           // 0..127
    const int a_k = (tid & 1) * 16;     // 0 / 16
    const int b_n = tid >> 2;           // 0..63
    const int b_k = (tid & 3) * 8;      // 0/8/16/24
    const int a_off = a_m * LDSTRIDE + a_k;
    const int b_off = b_n * LDSTRIDE + b_k;

    const float* xrow  = x + (size_t)(row0 + a_m) * DIM_H;
    const size_t wrow  = (size_t)(col0 + b_n);
    const int*   grow  = gcodes + wrow * DIM_H;
    const int*   urow  = ucodes + wrow * DIM_H;
    const float* gsrow = gscales + wrow * (DIM_H / NF_BLOCK);
    const float* usrow = uscales + wrow * (DIM_H / NF_BLOCK);

    v8f accG[4] = {};
    v8f accU[4] = {};

    Raw1 r = fetch1(xrow, grow, urow, gsrow, usrow, 0, a_k, b_k);
    int buf = 0;

    for (int k0 = 0; k0 < DIM_H; k0 += BK) {
        // ---- convert/dequant raw(k) -> LDS buffer[buf] ----
        {
            union { v8bf v; __bf16 e[8]; } h0, h1, bg, bu;
            #pragma unroll
            for (int j = 0; j < 4; ++j) {
                h0.e[j]     = f2bf(r.x0[j]);
                h0.e[4 + j] = f2bf(r.x1[j]);
                h1.e[j]     = f2bf(r.x2[j]);
                h1.e[4 + j] = f2bf(r.x3[j]);
                bg.e[j]     = f2bf(sNF4[r.g0[j] & 15] * r.gs);
                bg.e[4 + j] = f2bf(sNF4[r.g1[j] & 15] * r.gs);
                bu.e[j]     = f2bf(sNF4[r.u0[j] & 15] * r.us);
                bu.e[4 + j] = f2bf(sNF4[r.u1[j] & 15] * r.us);
            }
            *(v8bf*)&sA[buf][a_off]     = h0.v;
            *(v8bf*)&sA[buf][a_off + 8] = h1.v;
            *(v8bf*)&sG[buf][b_off]     = bg.v;
            *(v8bf*)&sU[buf][b_off]     = bu.v;
        }
        __syncthreads();

        // ---- issue loads for step k+1 (fly under the WMMAs below) ----
        if (k0 + BK < DIM_H) {
            r = fetch1(xrow, grow, urow, gsrow, usrow, k0 + BK, a_k, b_k);
            if (k0 + 2 * BK < DIM_H) {
                __builtin_prefetch(grow + k0 + 2 * BK + b_k, 0, 1);
                __builtin_prefetch(urow + k0 + 2 * BK + b_k, 0, 1);
            }
        }

        // ---- compute step k: 8 WMMAs per wave ----
        v16bf bG = load_frag(sG[buf], nw * 16 + ln, lh);
        v16bf bU = load_frag(sU[buf], nw * 16 + ln, lh);
        #pragma unroll
        for (int s = 0; s < 4; ++s) {
            v16bf a = load_frag(sA[buf], mw * 64 + s * 16 + ln, lh);
            accG[s] = __builtin_amdgcn_wmma_f32_16x16x32_bf16(
                false, a, false, bG, (short)0, accG[s], false, false);
            accU[s] = __builtin_amdgcn_wmma_f32_16x16x32_bf16(
                false, a, false, bU, (short)0, accU[s], false, false);
        }
        buf ^= 1;
    }

    // ---- epilogue: silu(gate)*up -> bf16 intermediate ----
    const int col = col0 + nw * 16 + ln;
    #pragma unroll
    for (int s = 0; s < 4; ++s) {
        #pragma unroll
        for (int v = 0; v < 8; ++v) {
            int m = mw * 64 + s * 16 + v + lh * 8;
            float g = accG[s][v];
            float u = accU[s][v];
            float si = g / (1.0f + __expf(-g));
            inter[(size_t)(row0 + m) * DIM_I + col] =
                __builtin_bit_cast(unsigned short, f2bf(si * u));
        }
    }
}

// =================================================================
// Kernel 2: out[bs,h] = inter @ Wd^T   (fp32 out)
// A operand staged by async global->LDS DMA (double-buffered)
// =================================================================
__global__ __launch_bounds__(256) void k_down(
    const unsigned short* __restrict__ inter,
    const int*   __restrict__ dcodes, const float* __restrict__ dscales,
    float* __restrict__ out)
{
    __shared__ alignas(16) unsigned short sA[2][BM * LDSTRIDE];
    __shared__ alignas(16) unsigned short sB[2][BN * LDSTRIDE];
    __shared__ float sNF4[16];

    const int tid = threadIdx.x;
    if (tid < 16) sNF4[tid] = NF4_TAB[tid];

    const int row0 = blockIdx.y * BM;
    const int col0 = blockIdx.x * BN;

    const int lane = tid & 31;
    const int wave = tid >> 5;
    const int mw = wave >> 2;
    const int nw = wave & 3;
    const int ln = lane & 15;
    const int lh = lane >> 4;

    const int a_m = tid >> 1;
    const int a_k = (tid & 1) * 16;
    const int b_n = tid >> 2;
    const int b_k = (tid & 3) * 8;
    const int a_off = a_m * LDSTRIDE + a_k;
    const int b_off = b_n * LDSTRIDE + b_k;

    const unsigned short* arow = inter + (size_t)(row0 + a_m) * DIM_I;
    const size_t wrow = (size_t)(col0 + b_n);
    const int*   drow = dcodes + wrow * DIM_I;
    const float* dsrow = dscales + wrow * (DIM_I / NF_BLOCK);

    v8f acc[4] = {};

    // ---- prologue: stage step 0 ----
    v4i c0 = *(const v4i*)(drow + b_k);
    v4i c1 = *(const v4i*)(drow + b_k + 4);
    float sc = dsrow[0];
#if USE_ASYNC
    __builtin_amdgcn_global_load_async_to_lds_b128((v4i*)(arow + a_k), (v4i*)&sA[0][a_off], 0, 0);
    __builtin_amdgcn_global_load_async_to_lds_b128((v4i*)(arow + a_k), (v4i*)&sA[0][a_off], 16, 0);
#else
    v4i ra0 = *(const v4i*)(arow + a_k);
    v4i ra1 = *(const v4i*)(arow + a_k + 8);
#endif
    int buf = 0;

    for (int k0 = 0; k0 < DIM_I; k0 += BK) {
        // ---- dequant B(k) -> sB[buf]; land A(k) ----
        {
            union { v8bf v; __bf16 e[8]; } bb;
            #pragma unroll
            for (int j = 0; j < 4; ++j) {
                bb.e[j]     = f2bf(sNF4[c0[j] & 15] * sc);
                bb.e[4 + j] = f2bf(sNF4[c1[j] & 15] * sc);
            }
            *(v8bf*)&sB[buf][b_off] = bb.v;
#if !USE_ASYNC
            *(v4i*)&sA[buf][a_off]     = ra0;
            *(v4i*)&sA[buf][a_off + 8] = ra1;
#endif
        }
#if USE_ASYNC
        __builtin_amdgcn_s_wait_asynccnt(0);   // A(k) DMA complete (per wave; barrier covers all)
#endif
        __syncthreads();

        // ---- issue step k+1 transfers (post-barrier: other buffer is free) ----
        if (k0 + BK < DIM_I) {
            const int kn = k0 + BK;
            c0 = *(const v4i*)(drow + kn + b_k);
            c1 = *(const v4i*)(drow + kn + b_k + 4);
            sc = dsrow[kn >> 6];
#if USE_ASYNC
            __builtin_amdgcn_global_load_async_to_lds_b128(
                (v4i*)(arow + kn + a_k), (v4i*)&sA[buf ^ 1][a_off], 0, 0);
            __builtin_amdgcn_global_load_async_to_lds_b128(
                (v4i*)(arow + kn + a_k), (v4i*)&sA[buf ^ 1][a_off], 16, 0);
#else
            ra0 = *(const v4i*)(arow + kn + a_k);
            ra1 = *(const v4i*)(arow + kn + a_k + 8);
#endif
            if (k0 + 2 * BK < DIM_I)
                __builtin_prefetch(drow + k0 + 2 * BK + b_k, 0, 1);
        }

        // ---- compute step k: 4 WMMAs per wave ----
        v16bf b = load_frag(sB[buf], nw * 16 + ln, lh);
        #pragma unroll
        for (int s = 0; s < 4; ++s) {
            v16bf a = load_frag(sA[buf], mw * 64 + s * 16 + ln, lh);
            acc[s] = __builtin_amdgcn_wmma_f32_16x16x32_bf16(
                false, a, false, b, (short)0, acc[s], false, false);
        }
        buf ^= 1;
    }

    const int col = col0 + nw * 16 + ln;
    #pragma unroll
    for (int s = 0; s < 4; ++s) {
        #pragma unroll
        for (int v = 0; v < 8; ++v) {
            int m = mw * 64 + s * 16 + v + lh * 8;
            out[(size_t)(row0 + m) * DIM_H + col] = acc[s][v];
        }
    }
}

// =================================================================
// Launch
// =================================================================
extern "C" void kernel_launch(void* const* d_in, const int* in_sizes, int n_in,
                              void* d_out, int out_size, void* d_ws, size_t ws_size,
                              hipStream_t stream) {
    (void)in_sizes; (void)n_in; (void)out_size; (void)ws_size;

    const float* x  = (const float*)d_in[0];
    const int*   gc = (const int*)  d_in[1];
    const float* gs = (const float*)d_in[2];
    const int*   uc = (const int*)  d_in[3];
    const float* us = (const float*)d_in[4];
    const int*   dc = (const int*)  d_in[5];
    const float* ds = (const float*)d_in[6];
    float* out = (float*)d_out;
    unsigned short* inter = (unsigned short*)d_ws;   // [8192, 11008] bf16 (180 MB)

    dim3 blk(256);
    dim3 g1(DIM_I / BN, DIM_BS / BM);   // 172 x 64
    k_gateup<<<g1, blk, 0, stream>>>(x, gc, gs, uc, us, inter);

    dim3 g2(DIM_H / BN, DIM_BS / BM);   // 64 x 64
    k_down<<<g2, blk, 0, stream>>>(inter, dc, ds, out);
}
